// MultiHeadSelfAttention_64733747085578
// MI455X (gfx1250) — matrix operand order repaired and checked by reading, not compiled, not run on hardware
//
#include <hip/hip_runtime.h>

#define EMB   1024
#define T_SEQ 2048
#define NHEAD 16
#define HDIM  64

typedef __attribute__((ext_vector_type(8)))  float  v8f;
typedef __attribute__((ext_vector_type(16))) __bf16 v16bf;

__device__ __forceinline__ unsigned short f32_to_bf16(float f) {
  unsigned int u = __float_as_uint(f);
  u += 0x7fffu + ((u >> 16) & 1u);   // round-to-nearest-even
  return (unsigned short)(u >> 16);
}

union Frag16 {
  uint4          q[2];
  unsigned short h[16];
  v16bf          v;
};

// ---------------------------------------------------------------------------
// fp32 -> bf16 conversion (straight copy)
// ---------------------------------------------------------------------------
__global__ void cvt_bf16_kernel(const float* __restrict__ in,
                                unsigned short* __restrict__ out, int n) {
  int i = blockIdx.x * blockDim.x + threadIdx.x;
  if (i < n) out[i] = f32_to_bf16(in[i]);
}

// ---------------------------------------------------------------------------
// fp32 [rows,cols] -> bf16 transposed [cols,rows]  (one-time weight prep)
// ---------------------------------------------------------------------------
__global__ void cvt_bf16_tr_kernel(const float* __restrict__ in,
                                   unsigned short* __restrict__ out,
                                   int rows, int cols) {
  int i = blockIdx.x * blockDim.x + threadIdx.x;
  if (i < rows * cols) {
    int r = i / cols, c = i - r * cols;
    out[(size_t)c * rows + r] = f32_to_bf16(in[i]);
  }
}

// ---------------------------------------------------------------------------
// QKV GEMM: [4096,1024](bf16) x W^T[3072,1024](bf16) + bias -> scatter to
// Q [B,H,T,Dh], K [B,H,T,Dh], V [B,H,Dh,T]  (all bf16).
// Block = 8 waves (2x4), block tile 64x128, wave tile 32x32, K-step 32.
// All LDS staging is b128; no scalar LDS stores.
// ---------------------------------------------------------------------------
__global__ __launch_bounds__(256) void qkv_gemm_kernel(
    const unsigned short* __restrict__ A,    // [4096,1024]
    const unsigned short* __restrict__ Bt,   // [3072,1024]  (W^T)
    const float* __restrict__ bias,          // [3072]
    unsigned short* __restrict__ qb,
    unsigned short* __restrict__ kb,
    unsigned short* __restrict__ vb) {
  __shared__ __align__(16) unsigned short As[64 * 32];   // [m][k]
  __shared__ __align__(16) unsigned short Bs[128 * 32];  // [n][k]

  const int tid  = threadIdx.x;
  const int wave = tid >> 5, lane = tid & 31;
  const int hl   = lane >> 4, lm = lane & 15;
  const int wm   = wave >> 2, wn = wave & 3;
  const int m0   = blockIdx.x * 64;
  const int n0   = blockIdx.y * 128;

  v8f acc[2][2] = {};

  const int arow = tid >> 2, acs = (tid & 3) * 8;        // A-tile mapping
  const int brow = tid >> 1, bcs = (tid & 1) * 16;       // B-tile mapping

  for (int k0 = 0; k0 < EMB; k0 += 32) {
    // A tile 64x32: one b128 per thread
    {
      const unsigned short* src = A + (size_t)(m0 + arow) * EMB + k0 + acs;
      *(uint4*)(As + arow * 32 + acs) = *(const uint4*)(src);
      __builtin_prefetch(src + 32, 0, 1);
    }
    // B tile 128x32 (already K-major): two b128 per thread
    {
      const unsigned short* src = Bt + (size_t)(n0 + brow) * EMB + k0 + bcs;
      *(uint4*)(Bs + brow * 32 + bcs)     = *(const uint4*)(src);
      *(uint4*)(Bs + brow * 32 + bcs + 8) = *(const uint4*)(src + 8);
      __builtin_prefetch(src + 32, 0, 1);
    }
    __syncthreads();

    Frag16 afr[2], bfr[2];
    #pragma unroll
    for (int mi = 0; mi < 2; ++mi) {
      const unsigned short* p = As + (wm * 32 + mi * 16 + lm) * 32;
      afr[mi].q[0] = *(const uint4*)(p + hl * 8);
      afr[mi].q[1] = *(const uint4*)(p + 16 + hl * 8);
    }
    #pragma unroll
    for (int ni = 0; ni < 2; ++ni) {
      const unsigned short* p = Bs + (wn * 32 + ni * 16 + lm) * 32 + hl * 8;
      bfr[ni].q[0] = *(const uint4*)(p);
      bfr[ni].q[1] = *(const uint4*)(p + 16);
    }
    #pragma unroll
    for (int mi = 0; mi < 2; ++mi)
      #pragma unroll
      for (int ni = 0; ni < 2; ++ni)
        acc[mi][ni] = __builtin_amdgcn_wmma_f32_16x16x32_bf16(
            false, afr[mi].v, false, bfr[ni].v, (short)0, acc[mi][ni],
            false, false);
    __syncthreads();
  }

  // scatter with fused bias
  #pragma unroll
  for (int mi = 0; mi < 2; ++mi) {
    int gm = m0 + wm * 32 + mi * 16 + hl * 8;        // rows gm..gm+7, same b
    int b  = gm >> 11, t0 = gm & 2047;
    #pragma unroll
    for (int ni = 0; ni < 2; ++ni) {
      int gn = n0 + wn * 32 + ni * 16 + lm;
      float bv  = bias[gn];
      int which = gn >> 10;
      int e = gn & 1023;
      int h = e >> 6, d = e & 63;
      if (which == 2) {
        // V in [B,H,Dh,T]: 8 consecutive t -> one b128 store
        unsigned short tmp[8] __attribute__((aligned(16)));
        #pragma unroll
        for (int r = 0; r < 8; ++r)
          tmp[r] = f32_to_bf16(acc[mi][ni][r] + bv);
        *(uint4*)(vb + (((size_t)(b * NHEAD + h)) * HDIM + d) * T_SEQ + t0) =
            *(const uint4*)tmp;
      } else {
        unsigned short* dst = (which == 0) ? qb : kb;
        #pragma unroll
        for (int r = 0; r < 8; ++r)
          dst[(((size_t)(b * NHEAD + h)) * T_SEQ + (t0 + r)) * HDIM + d] =
              f32_to_bf16(acc[mi][ni][r] + bv);
      }
    }
  }
}

// ---------------------------------------------------------------------------
// Flash attention: grid (T/128, B*H). 8 waves per block, 16 q-rows per wave.
// Key tiles of 32, online softmax, all matmuls via bf16 WMMA w/ f32 accum.
// V arrives pre-transposed [B,H,Dh,T] so staging is pure b128.
// ---------------------------------------------------------------------------
__global__ __launch_bounds__(256) void attn_kernel(
    const unsigned short* __restrict__ Q,    // [B,H,T,Dh]
    const unsigned short* __restrict__ K,    // [B,H,T,Dh]
    const unsigned short* __restrict__ V,    // [B,H,Dh,T]
    unsigned short* __restrict__ O) {        // [B,T,E] bf16
  __shared__ __align__(16) unsigned short Ks[32 * 64];      // [key][d]
  __shared__ __align__(16) unsigned short Vst[64 * 32];     // [d][key]
  __shared__ __align__(16) unsigned short Ps[8 * 16 * 32];  // per-wave P

  const int tid  = threadIdx.x;
  const int wave = tid >> 5, lane = tid & 31;
  const int hl   = lane >> 4, lm = lane & 15;
  const int bh   = blockIdx.y;                 // b*16+h
  const int q0   = blockIdx.x * 128 + wave * 16;
  const size_t base = (size_t)bh * T_SEQ * HDIM;   // same span for Q/K/V

  // Q fragments for d-blocks 0..31 and 32..63 (loaded once)
  Frag16 qf[2];
  {
    const unsigned short* qp = Q + base + (size_t)(q0 + lm) * HDIM;
    #pragma unroll
    for (int db = 0; db < 2; ++db) {
      qf[db].q[0] = *(const uint4*)(qp + db * 32 + hl * 8);
      qf[db].q[1] = *(const uint4*)(qp + db * 32 + 16 + hl * 8);
    }
  }

  float m_run[8], l_run[8];
  #pragma unroll
  for (int r = 0; r < 8; ++r) { m_run[r] = -1e30f; l_run[r] = 0.f; }
  v8f oacc[4] = {};
  const float scale = 0.125f;                  // 1/sqrt(64)
  unsigned short* myP = Ps + wave * (16 * 32);

  const int krow = tid >> 3, kcs = (tid & 7) * 8;   // K coop mapping
  const int vrow = tid >> 2, vcs = (tid & 3) * 8;   // V coop mapping

  for (int kt0 = 0; kt0 < T_SEQ; kt0 += 32) {
    // cooperative staging, all b128
    {
      const unsigned short* ksrc =
          K + base + (size_t)(kt0 + krow) * HDIM + kcs;
      *(uint4*)(Ks + krow * 64 + kcs) = *(const uint4*)(ksrc);
      const unsigned short* vsrc =
          V + base + (size_t)vrow * T_SEQ + kt0 + vcs;
      *(uint4*)(Vst + vrow * 32 + vcs) = *(const uint4*)(vsrc);
      __builtin_prefetch(ksrc + 32 * HDIM, 0, 1);
      __builtin_prefetch(vsrc + 32, 0, 1);
    }
    __syncthreads();

    // S = Q * K^T for 32 keys: two 16x16 C tiles, 2 WMMAs each (d halves)
    v8f s[2] = {};
    #pragma unroll
    for (int kt = 0; kt < 2; ++kt) {
      #pragma unroll
      for (int db = 0; db < 2; ++db) {
        Frag16 kf;
        const unsigned short* p = Ks + (kt * 16 + lm) * 64 + db * 32 + hl * 8;
        kf.q[0] = *(const uint4*)(p);
        kf.q[1] = *(const uint4*)(p + 16);
        s[kt] = __builtin_amdgcn_wmma_f32_16x16x32_bf16(
            false, qf[db].v, false, kf.v, (short)0, s[kt], false, false);
      }
    }

    // online softmax per row (8 rows per lane-half; reductions over 16 lanes)
    float alpha[8];
    #pragma unroll
    for (int r = 0; r < 8; ++r) {
      float s0 = s[0][r] * scale;
      float s1 = s[1][r] * scale;
      float vmax = fmaxf(s0, s1);
      #pragma unroll
      for (int msk = 1; msk <= 8; msk <<= 1)
        vmax = fmaxf(vmax, __shfl_xor(vmax, msk, 32));
      float mnew = fmaxf(m_run[r], vmax);
      float p0 = __expf(s0 - mnew);
      float p1 = __expf(s1 - mnew);
      float psum = p0 + p1;
      #pragma unroll
      for (int msk = 1; msk <= 8; msk <<= 1)
        psum += __shfl_xor(psum, msk, 32);
      alpha[r] = __expf(m_run[r] - mnew);
      l_run[r] = l_run[r] * alpha[r] + psum;
      m_run[r] = mnew;
      int mrow = hl * 8 + r;                    // C-layout row owned by lane
      myP[mrow * 32 + lm]      = f32_to_bf16(p0);
      myP[mrow * 32 + 16 + lm] = f32_to_bf16(p1);
    }
    #pragma unroll
    for (int j = 0; j < 4; ++j)
      #pragma unroll
      for (int r = 0; r < 8; ++r)
        oacc[j][r] *= alpha[r];
    __syncthreads();

    // O += P * V : P re-read in A-fragment layout from LDS
    Frag16 pf;
    {
      const unsigned short* p = myP + lm * 32;
      pf.q[0] = *(const uint4*)(p + hl * 8);
      pf.q[1] = *(const uint4*)(p + 16 + hl * 8);
    }
    #pragma unroll
    for (int j = 0; j < 4; ++j) {
      Frag16 vf;
      const unsigned short* p = Vst + (j * 16 + lm) * 32 + hl * 8;
      vf.q[0] = *(const uint4*)(p);
      vf.q[1] = *(const uint4*)(p + 16);
      oacc[j] = __builtin_amdgcn_wmma_f32_16x16x32_bf16(
          false, pf.v, false, vf.v, (short)0, oacc[j], false, false);
    }
    __syncthreads();
  }

  // normalize + write bf16 into [B,T,E]
  int b = bh >> 4, h = bh & 15;
  #pragma unroll
  for (int r = 0; r < 8; ++r) {
    float inv = 1.0f / l_run[r];
    int t = q0 + hl * 8 + r;
    unsigned short* op = O + ((size_t)b * T_SEQ + t) * EMB + h * HDIM;
    #pragma unroll
    for (int j = 0; j < 4; ++j)
      op[j * 16 + lm] = f32_to_bf16(oacc[j][r] * inv);
  }
}

// ---------------------------------------------------------------------------
// Output projection: [4096,1024](bf16) x W^T[1024,1024](bf16) + bias -> f32
// ---------------------------------------------------------------------------
__global__ __launch_bounds__(256) void fc_gemm_kernel(
    const unsigned short* __restrict__ A,    // [4096,1024] attn output
    const unsigned short* __restrict__ Bt,   // [1024,1024]  (W^T)
    const float* __restrict__ bias,
    float* __restrict__ C) {
  __shared__ __align__(16) unsigned short As[64 * 32];
  __shared__ __align__(16) unsigned short Bs[128 * 32];

  const int tid  = threadIdx.x;
  const int wave = tid >> 5, lane = tid & 31;
  const int hl   = lane >> 4, lm = lane & 15;
  const int wm   = wave >> 2, wn = wave & 3;
  const int m0   = blockIdx.x * 64;
  const int n0   = blockIdx.y * 128;

  v8f acc[2][2] = {};

  const int arow = tid >> 2, acs = (tid & 3) * 8;
  const int brow = tid >> 1, bcs = (tid & 1) * 16;

  for (int k0 = 0; k0 < EMB; k0 += 32) {
    {
      const unsigned short* src = A + (size_t)(m0 + arow) * EMB + k0 + acs;
      *(uint4*)(As + arow * 32 + acs) = *(const uint4*)(src);
      __builtin_prefetch(src + 32, 0, 1);
    }
    {
      const unsigned short* src = Bt + (size_t)(n0 + brow) * EMB + k0 + bcs;
      *(uint4*)(Bs + brow * 32 + bcs)     = *(const uint4*)(src);
      *(uint4*)(Bs + brow * 32 + bcs + 8) = *(const uint4*)(src + 8);
      __builtin_prefetch(src + 32, 0, 1);
    }
    __syncthreads();

    Frag16 afr[2], bfr[2];
    #pragma unroll
    for (int mi = 0; mi < 2; ++mi) {
      const unsigned short* p = As + (wm * 32 + mi * 16 + lm) * 32;
      afr[mi].q[0] = *(const uint4*)(p + hl * 8);
      afr[mi].q[1] = *(const uint4*)(p + 16 + hl * 8);
    }
    #pragma unroll
    for (int ni = 0; ni < 2; ++ni) {
      const unsigned short* p = Bs + (wn * 32 + ni * 16 + lm) * 32 + hl * 8;
      bfr[ni].q[0] = *(const uint4*)(p);
      bfr[ni].q[1] = *(const uint4*)(p + 16);
    }
    #pragma unroll
    for (int mi = 0; mi < 2; ++mi)
      #pragma unroll
      for (int ni = 0; ni < 2; ++ni)
        acc[mi][ni] = __builtin_amdgcn_wmma_f32_16x16x32_bf16(
            false, afr[mi].v, false, bfr[ni].v, (short)0, acc[mi][ni],
            false, false);
    __syncthreads();
  }

  #pragma unroll
  for (int mi = 0; mi < 2; ++mi) {
    int gm = m0 + wm * 32 + mi * 16 + hl * 8;
    #pragma unroll
    for (int ni = 0; ni < 2; ++ni) {
      int gn = n0 + wn * 32 + ni * 16 + lm;
      float bv = bias[gn];
      #pragma unroll
      for (int r = 0; r < 8; ++r)
        C[(size_t)(gm + r) * EMB + gn] = acc[mi][ni][r] + bv;
    }
  }
}

// ---------------------------------------------------------------------------
extern "C" void kernel_launch(void* const* d_in, const int* in_sizes, int n_in,
                              void* d_out, int out_size, void* d_ws,
                              size_t ws_size, hipStream_t stream) {
  const float* x     = (const float*)d_in[0];
  const float* w_qkv = (const float*)d_in[1];
  const float* b_qkv = (const float*)d_in[2];
  const float* w_fc  = (const float*)d_in[3];
  const float* b_fc  = (const float*)d_in[4];
  float* out = (float*)d_out;

  unsigned short* ws    = (unsigned short*)d_ws;
  unsigned short* xb    = ws;                       // 4,194,304
  unsigned short* wqkvT = xb + 4194304;             // 3,145,728  [3072,1024]
  unsigned short* wfcT  = wqkvT + 3145728;          // 1,048,576  [1024,1024]
  unsigned short* qbuf  = wfcT + 1048576;           // 4,194,304  [B,H,T,Dh]
  unsigned short* kbuf  = qbuf + 4194304;           // 4,194,304  [B,H,T,Dh]
  unsigned short* vbuf  = kbuf + 4194304;           // 4,194,304  [B,H,Dh,T]
  unsigned short* ao    = vbuf + 4194304;           // 4,194,304  [B,T,E]

  cvt_bf16_kernel<<<4194304 / 256, 256, 0, stream>>>(x, xb, 4194304);
  cvt_bf16_tr_kernel<<<3145728 / 256, 256, 0, stream>>>(w_qkv, wqkvT,
                                                        1024, 3072);
  cvt_bf16_tr_kernel<<<1048576 / 256, 256, 0, stream>>>(w_fc, wfcT,
                                                        1024, 1024);

  qkv_gemm_kernel<<<dim3(64, 24), 256, 0, stream>>>(xb, wqkvT, b_qkv,
                                                    qbuf, kbuf, vbuf);
  attn_kernel<<<dim3(16, 32), 256, 0, stream>>>(qbuf, kbuf, vbuf, ao);
  fc_gemm_kernel<<<dim3(64, 8), 256, 0, stream>>>(ao, wfcT, b_fc, out);
}
